// CausalSelfAttention_146028888627
// MI455X (gfx1250) — matrix operand order repaired and checked by reading, not compiled, not run on hardware
//
#include <hip/hip_runtime.h>
#include <stdint.h>

// ---------------------------------------------------------------------------
// Problem constants (B=2, T=2048, C=1024, H=16, Dh=64)
// ---------------------------------------------------------------------------
#define B_   2
#define T_   2048
#define C_   1024
#define H_   16
#define DH_  64
#define TC_  (T_ * C_)        // 2097152 elements per batch
#define HST_ (T_ * DH_)       // 131072 elements per head view

typedef __attribute__((ext_vector_type(16))) __bf16   v16bf;
typedef __attribute__((ext_vector_type(8)))  float    v8f;
typedef __attribute__((ext_vector_type(4)))  unsigned tdm_g0_t;
typedef __attribute__((ext_vector_type(8)))  int      tdm_g1_t;
typedef __attribute__((ext_vector_type(4)))  int      tdm_g2_t;

union AFrag { v16bf h; uint4 q[2]; };

__device__ __forceinline__ unsigned short f2bf(float f) {
  union { float f; unsigned u; } v; v.f = f;
  unsigned r = v.u + 0x7FFFu + ((v.u >> 16) & 1u);   // round-to-nearest-even
  return (unsigned short)(r >> 16);
}

// ---------------------------------------------------------------------------
// TDM: async 2D tile load global -> LDS (CDNA5 Tensor Data Mover, ISA ch.8).
// Descriptor built per D# spec: group0 = {count, lds_addr, global_addr, type=2},
// group1 = {data_size=2B, tensor dims, tile dims, dim0 stride}. 2D only, so
// groups 2/3 are zero. Completion tracked with TENSORcnt.
// ---------------------------------------------------------------------------
#if __has_builtin(__builtin_amdgcn_tensor_load_to_lds)
#define ATH_HAS_TDM 1
#else
#define ATH_HAS_TDM 0
#endif

__device__ __forceinline__ void tdm_load_2d(unsigned lds_off, const void* gptr,
                                            unsigned tensor_d0, unsigned tensor_d1,
                                            unsigned stride0,
                                            unsigned tile_d0, unsigned tile_d1) {
#if ATH_HAS_TDM
  unsigned long long ga = (unsigned long long)(uintptr_t)gptr;
  tdm_g0_t g0;
  g0.x = 1u;                                             // count=1 valid descriptor
  g0.y = lds_off;                                        // LDS byte address
  g0.z = (unsigned)(ga & 0xFFFFFFFFu);                   // global_addr[31:0]
  g0.w = (unsigned)((ga >> 32) & 0x01FFFFFFu) | (2u << 30); // global_addr[56:32] | type=2
  tdm_g1_t g1;
  g1[0] = (int)(1u << 16);                               // data_size=1 -> 2 bytes/elem
  g1[1] = (int)((tensor_d0 & 0xFFFFu) << 16);            // tensor_dim0[15:0]
  g1[2] = (int)(((tensor_d0 >> 16) & 0xFFFFu) | ((tensor_d1 & 0xFFFFu) << 16));
  g1[3] = (int)(((tensor_d1 >> 16) & 0xFFFFu) | ((tile_d0 & 0xFFFFu) << 16));
  g1[4] = (int)(tile_d1 & 0xFFFFu);                      // tile_dim1, tile_dim2=0
  g1[5] = (int)stride0;                                  // tensor_dim0_stride[31:0]
  g1[6] = 0;
  g1[7] = 0;
  tdm_g2_t gz = (tdm_g2_t)0;
#if __clang_major__ >= 23
  __builtin_amdgcn_tensor_load_to_lds(g0, g1, gz, gz, (tdm_g1_t)0, 0);
#else
  __builtin_amdgcn_tensor_load_to_lds(g0, g1, gz, gz, 0);
#endif
#endif
}

__device__ __forceinline__ void wait_tensorcnt_le2() {
#if ATH_HAS_TDM
  __builtin_amdgcn_s_wait_tensorcnt(2);
#endif
}
__device__ __forceinline__ void wait_tensorcnt_le0() {
#if ATH_HAS_TDM
  __builtin_amdgcn_s_wait_tensorcnt(0);
#endif
}

// Stage a tile_rows x tile_cols bf16 tile (row stride stride0 elems) into LDS.
// TDM path: wave 0 issues async DMA. Fallback: cooperative synchronous copy.
__device__ __forceinline__ void stage2d(char* smem, unsigned lds_off,
                                        const unsigned short* g, unsigned stride0,
                                        unsigned tile_rows, unsigned tile_cols,
                                        unsigned tensor_d0, unsigned tensor_d1,
                                        int wid, int tid) {
#if ATH_HAS_TDM
  (void)smem; (void)tid;
  if (wid == 0) tdm_load_2d(lds_off, g, tensor_d0, tensor_d1, stride0, tile_cols, tile_rows);
#else
  uint4* dst = (uint4*)(smem + lds_off);
  unsigned cpr = tile_cols >> 3;                 // uint4 chunks per row
  unsigned total = tile_rows * cpr;
  for (unsigned i = (unsigned)tid; i < total; i += 256u) {
    unsigned r = i / cpr, c = (i % cpr) << 3;
    dst[i] = *(const uint4*)(g + (size_t)r * stride0 + c);
  }
#endif
}

// ---------------------------------------------------------------------------
// Kernel 1: fp32 -> bf16 conversion
// ---------------------------------------------------------------------------
__global__ void cvt_bf16_kernel(const float* __restrict__ in,
                                unsigned short* __restrict__ out, int n) {
  int i = blockIdx.x * 256 + threadIdx.x;
  if (i < n) out[i] = f2bf(in[i]);
}

// ---------------------------------------------------------------------------
// Kernel 2: QKV projection  out[t][o] = sum_c x[t][c] * W[o][c]
// Both operands have K contiguous (NT GEMM) -> natural WMMA A/B layouts.
// Block tile 128(M) x 128(N), K step 32. 8 waves: 4 along M (32 rows each),
// 2 along N (64 cols each). Per wave: 2x4 = 8 v_wmma_f32_16x16x32_bf16 / step.
// LDS double buffered: A 8KB + B 8KB per buffer (32KB total), TDM staged.
// ---------------------------------------------------------------------------
__global__ __launch_bounds__(256) void qkv_gemm_kernel(
    const unsigned short* __restrict__ xb,
    const unsigned short* __restrict__ Wqb,
    const unsigned short* __restrict__ Wkb,
    const unsigned short* __restrict__ Wvb,
    unsigned short* __restrict__ Qo,
    unsigned short* __restrict__ Ko,
    unsigned short* __restrict__ Vo) {
  extern __shared__ char smem[];
  const int m0 = blockIdx.x * 128;
  const int n0 = blockIdx.y * 128;
  const int which = blockIdx.z;
  const unsigned short* W = (which == 0) ? Wqb : ((which == 1) ? Wkb : Wvb);
  unsigned short* Out = (which == 0) ? Qo : ((which == 1) ? Ko : Vo);

  const int tid = threadIdx.x;
  const int wid = tid >> 5, lane = tid & 31;
  const int wm = (wid & 3) * 32;
  const int wn = (wid >> 2) * 64;
  const int ml = lane & 15, kh = lane >> 4;

  v8f acc[2][4];
  for (int i = 0; i < 2; i++)
    for (int j = 0; j < 4; j++) acc[i][j] = (v8f)0.0f;

  auto Aoff = [](int buf) -> unsigned { return (unsigned)buf * 16384u; };
  auto Boff = [](int buf) -> unsigned { return (unsigned)buf * 16384u + 8192u; };

  const int NK = C_ / 32;  // 32 k-steps
  stage2d(smem, Aoff(0), xb + (size_t)m0 * C_, C_, 128, 32, C_, B_ * T_, wid, tid);
  stage2d(smem, Boff(0), W + (size_t)n0 * C_, C_, 128, 32, C_, C_, wid, tid);

  for (int ks = 0; ks < NK; ks++) {
    const int cur = ks & 1;
    if (ks + 1 < NK) {
      const int k1 = (ks + 1) * 32;
      stage2d(smem, Aoff(cur ^ 1), xb + (size_t)m0 * C_ + k1, C_, 128, 32, C_, B_ * T_, wid, tid);
      stage2d(smem, Boff(cur ^ 1), W + (size_t)n0 * C_ + k1, C_, 128, 32, C_, C_, wid, tid);
      wait_tensorcnt_le2();   // tiles for step ks are complete; next pair in flight
    } else {
      wait_tensorcnt_le0();
    }
    __syncthreads();

    const unsigned short* lA = (const unsigned short*)(smem + Aoff(cur));
    const unsigned short* lB = (const unsigned short*)(smem + Boff(cur));

    // A fragment (16x32 bf16): lane m holds K 0-7 & 16-23 (lo half) / 8-15 & 24-31
    AFrag a[2];
    for (int i = 0; i < 2; i++) {
      const unsigned short* p = lA + (wm + i * 16 + ml) * 32;
      a[i].q[0] = *(const uint4*)(p + kh * 8);
      a[i].q[1] = *(const uint4*)(p + 16 + kh * 8);
    }
    // B fragment (32x16 bf16): lane n holds K 0-7 & 16-23 (lo half) / 8-15 & 24-31
    for (int j = 0; j < 4; j++) {
      const unsigned short* p = lB + (wn + j * 16 + ml) * 32;
      AFrag b;
      b.q[0] = *(const uint4*)(p + kh * 8);
      b.q[1] = *(const uint4*)(p + 16 + kh * 8);
      for (int i = 0; i < 2; i++)
        acc[i][j] = __builtin_amdgcn_wmma_f32_16x16x32_bf16(
            false, a[i].h, false, b.h, (short)0, acc[i][j], false, false);
    }
    __syncthreads();
  }

  // D layout: VGPR r, lane l -> row = r + 8*(l>=16), col = l%16
  for (int i = 0; i < 2; i++)
    for (int j = 0; j < 4; j++)
      for (int r = 0; r < 8; r++) {
        int row = m0 + wm + i * 16 + r + 8 * kh;
        int col = n0 + wn + j * 16 + ml;
        Out[(size_t)row * C_ + col] = f2bf(acc[i][j][r]);
      }
}

// ---------------------------------------------------------------------------
// Kernel 3: per-head V transpose (P@V needs V^T as WMMA B operand:
// lane n = d must hold contiguous key-dim values)
// ---------------------------------------------------------------------------
__global__ void vtrans_kernel(const unsigned short* __restrict__ V,
                              unsigned short* __restrict__ Vt) {
  size_t g = (size_t)blockIdx.x * 256 + threadIdx.x;  // over B*H*T*Dh = 4194304
  unsigned bh = (unsigned)(g >> 17);
  unsigned rem = (unsigned)(g & 131071u);
  unsigned t = rem >> 6, d = rem & 63u;
  Vt[((size_t)bh << 17) + (size_t)d * T_ + t] = V[g];
}

// ---------------------------------------------------------------------------
// Kernel 4: causal flash attention per (b, h, 128-row q-block).
// 8 waves x 16 q-rows. Key blocks of 64, K & V^T tiles TDM double buffered in
// LDS. Online softmax: row stats live per-lane (8 rows each), reduced across
// the 16 lanes of each half-wave with __shfl_xor (wave32). P goes through a
// per-wave LDS slab to convert WMMA D layout -> A layout. 16 WMMAs / key blk.
// ---------------------------------------------------------------------------
__global__ __launch_bounds__(256) void attn_kernel(
    const unsigned short* __restrict__ Qg,
    const unsigned short* __restrict__ Kg,
    const unsigned short* __restrict__ Vtg,
    float* __restrict__ out) {
  extern __shared__ char smem[];
  const int qb = blockIdx.x, h = blockIdx.y, b = blockIdx.z;
  const size_t hoff = (size_t)b * TC_ + (size_t)h * HST_;
  const unsigned short* Qh = Qg + hoff;
  const unsigned short* Kh = Kg + hoff;
  const unsigned short* Vth = Vtg + hoff;
  float* Oh = out + hoff;

  const int tid = threadIdx.x, wid = tid >> 5, lane = tid & 31;
  const int ml = lane & 15, hh = lane >> 4;
  const int q0w = qb * 128 + wid * 16;

  auto Koff = [](int buf) -> unsigned { return (unsigned)buf * 16384u; };
  auto Voff = [](int buf) -> unsigned { return (unsigned)buf * 16384u + 8192u; };
  unsigned short* lsP = (unsigned short*)(smem + 32768u + (unsigned)wid * 2048u);

  // Q fragments for Dh=64: two 16x32 A-fragments, kept in registers all along
  AFrag qf[2];
  for (int f = 0; f < 2; f++) {
    const unsigned short* p = Qh + (size_t)(q0w + ml) * DH_ + f * 32;
    qf[f].q[0] = *(const uint4*)(p + hh * 8);
    qf[f].q[1] = *(const uint4*)(p + 16 + hh * 8);
  }

  v8f o[4];
  float mrow[8], lrow[8];
  for (int j = 0; j < 4; j++) o[j] = (v8f)0.0f;
  for (int r = 0; r < 8; r++) { mrow[r] = -__builtin_inff(); lrow[r] = 0.0f; }

  const int nkb = 2 * (qb + 1);       // causal: keys < (qb+1)*128
  const float scale = 0.125f;         // 1/sqrt(Dh)

  stage2d(smem, Koff(0), Kh, DH_, 64, DH_, DH_, T_, wid, tid);
  stage2d(smem, Voff(0), Vth, T_, DH_, 64, T_, DH_, wid, tid);

  for (int kb = 0; kb < nkb; kb++) {
    const int cur = kb & 1;
    if (kb + 1 < nkb) {
      stage2d(smem, Koff(cur ^ 1), Kh + (size_t)(kb + 1) * 64 * DH_, DH_, 64, DH_, DH_, T_, wid, tid);
      stage2d(smem, Voff(cur ^ 1), Vth + (size_t)(kb + 1) * 64, T_, DH_, 64, T_, DH_, wid, tid);
      wait_tensorcnt_le2();
    } else {
      wait_tensorcnt_le0();
    }
    __syncthreads();

    const unsigned short* lK = (const unsigned short*)(smem + Koff(cur));
    const unsigned short* lV = (const unsigned short*)(smem + Voff(cur));

    // ---- S = Q * K^T (16 q-rows x 64 keys), 8 WMMAs ----
    v8f s[4];
    for (int j = 0; j < 4; j++) s[j] = (v8f)0.0f;
    for (int kk = 0; kk < 2; kk++)
      for (int j = 0; j < 4; j++) {
        AFrag bk;
        const unsigned short* p = lK + (j * 16 + ml) * DH_ + kk * 32;
        bk.q[0] = *(const uint4*)(p + hh * 8);
        bk.q[1] = *(const uint4*)(p + 16 + hh * 8);
        s[j] = __builtin_amdgcn_wmma_f32_16x16x32_bf16(
            false, qf[kk].h, false, bk.h, (short)0, s[j], false, false);
      }

    // ---- scale + causal mask ----
    const int kbase = kb * 64;
    const bool diag = (kbase + 63) > q0w;   // wave-uniform
    for (int j = 0; j < 4; j++)
      for (int r = 0; r < 8; r++) {
        float v = s[j][r] * scale;
        if (diag) {
          int kgl = kbase + j * 16 + ml;
          int qgl = q0w + r + 8 * hh;
          if (kgl > qgl) v = -__builtin_inff();
        }
        s[j][r] = v;
      }

    // ---- online softmax: row max across the 16 lanes holding each row ----
    float mnew[8];
    for (int r = 0; r < 8; r++)
      mnew[r] = fmaxf(fmaxf(s[0][r], s[1][r]), fmaxf(s[2][r], s[3][r]));
    for (int msk = 1; msk < 16; msk <<= 1)
      for (int r = 0; r < 8; r++)
        mnew[r] = fmaxf(mnew[r], __shfl_xor(mnew[r], msk, 32));

    float alpha[8], lnew[8];
    for (int r = 0; r < 8; r++) {
      float mi = fmaxf(mrow[r], mnew[r]);
      alpha[r] = __expf(mrow[r] - mi);   // first iter: exp(-inf)=0
      mrow[r] = mi;
      lnew[r] = 0.0f;
    }

    // ---- P = exp(S - m), D-layout -> per-wave LDS slab (A-layout source) ----
    for (int j = 0; j < 4; j++)
      for (int r = 0; r < 8; r++) {
        float p = __expf(s[j][r] - mrow[r]);
        lnew[r] += p;
        lsP[(r + 8 * hh) * 64 + j * 16 + ml] = f2bf(p);
      }
    for (int msk = 1; msk < 16; msk <<= 1)
      for (int r = 0; r < 8; r++) lnew[r] += __shfl_xor(lnew[r], msk, 32);
    for (int r = 0; r < 8; r++) lrow[r] = lrow[r] * alpha[r] + lnew[r];
    for (int j = 0; j < 4; j++)
      for (int r = 0; r < 8; r++) o[j][r] *= alpha[r];

    // ---- O += P * V (B operand = V^T tile), 8 WMMAs ----
    for (int kk = 0; kk < 2; kk++) {
      AFrag ap;
      const unsigned short* pp = lsP + ml * 64 + kk * 32;
      ap.q[0] = *(const uint4*)(pp + hh * 8);
      ap.q[1] = *(const uint4*)(pp + 16 + hh * 8);
      for (int j = 0; j < 4; j++) {
        AFrag bv;
        const unsigned short* p = lV + (j * 16 + ml) * 64 + kk * 32;
        bv.q[0] = *(const uint4*)(p + hh * 8);
        bv.q[1] = *(const uint4*)(p + 16 + hh * 8);
        o[j] = __builtin_amdgcn_wmma_f32_16x16x32_bf16(
            false, ap.h, false, bv.h, (short)0, o[j], false, false);
      }
    }
    __syncthreads();
  }

  // ---- normalize and store fp32 (flat layout == reference reshape) ----
  for (int j = 0; j < 4; j++)
    for (int r = 0; r < 8; r++) {
      int qgl = q0w + r + 8 * hh;
      int d = j * 16 + ml;
      Oh[(size_t)qgl * DH_ + d] = o[j][r] / lrow[r];
    }
}

// ---------------------------------------------------------------------------
// Host launcher. Workspace layout (bf16 buffers, 46 MB total):
//   [0,8M)   xb       [8M,10M) Wqb   [10M,12M) Wkb   [12M,14M) Wvb
//   [14M,22M) Q   [22M,30M) K   [30M,38M) V   [38M,46M) V^T
// ---------------------------------------------------------------------------
extern "C" void kernel_launch(void* const* d_in, const int* in_sizes, int n_in,
                              void* d_out, int out_size, void* d_ws, size_t ws_size,
                              hipStream_t stream) {
  (void)in_sizes; (void)n_in; (void)out_size; (void)ws_size;
  const float* x  = (const float*)d_in[0];
  const float* Wq = (const float*)d_in[1];
  const float* Wk = (const float*)d_in[2];
  const float* Wv = (const float*)d_in[3];

  char* ws = (char*)d_ws;
  const size_t MB = 1u << 20;
  unsigned short* xb  = (unsigned short*)(ws + 0 * MB);
  unsigned short* Wqb = (unsigned short*)(ws + 8 * MB);
  unsigned short* Wkb = (unsigned short*)(ws + 10 * MB);
  unsigned short* Wvb = (unsigned short*)(ws + 12 * MB);
  unsigned short* Qb  = (unsigned short*)(ws + 14 * MB);
  unsigned short* Kb  = (unsigned short*)(ws + 22 * MB);
  unsigned short* Vb  = (unsigned short*)(ws + 30 * MB);
  unsigned short* Vtb = (unsigned short*)(ws + 38 * MB);

  cvt_bf16_kernel<<<(B_ * T_ * C_) / 256, 256, 0, stream>>>(x, xb, B_ * T_ * C_);
  cvt_bf16_kernel<<<(C_ * C_) / 256, 256, 0, stream>>>(Wq, Wqb, C_ * C_);
  cvt_bf16_kernel<<<(C_ * C_) / 256, 256, 0, stream>>>(Wk, Wkb, C_ * C_);
  cvt_bf16_kernel<<<(C_ * C_) / 256, 256, 0, stream>>>(Wv, Wvb, C_ * C_);

  qkv_gemm_kernel<<<dim3((B_ * T_) / 128, C_ / 128, 3), 256, 32768, stream>>>(
      xb, Wqb, Wkb, Wvb, Qb, Kb, Vb);

  vtrans_kernel<<<(B_ * H_ * T_ * DH_) / 256, 256, 0, stream>>>(Vb, Vtb);

  attn_kernel<<<dim3(T_ / 128, H_, B_), 256, 49152, stream>>>(
      Qb, Kb, Vtb, (float*)d_out);
}
